// GatedPropagation_54133767799114
// MI455X (gfx1250) — matrix operand order repaired and checked by reading, not compiled, not run on hardware
//
#include <hip/hip_runtime.h>

#define LSEQ 1024
#define BS 8
#define DQK 256
#define DVU 256
#define NH 8
#define DEXP 512
#define HD 64
#define DATT 32
#define HALF_IN 128
#define MROWS (LSEQ * BS) /* 8192 */

typedef __attribute__((ext_vector_type(16))) __bf16 v16bf;
typedef __attribute__((ext_vector_type(8)))  __bf16 v8bf;
typedef __attribute__((ext_vector_type(8)))  float  v8f;
typedef __attribute__((ext_vector_type(4)))  float  f4v;
typedef __attribute__((ext_vector_type(4)))  unsigned int u4v;

__device__ __forceinline__ unsigned short f2us(float f) {
  __bf16 b = (__bf16)f;  // hardware RNE convert
  unsigned short u; __builtin_memcpy(&u, &b, 2); return u;
}

// ---- WMMA fragment loaders (CDNA5 ISA 7.12.2, wave32) ----------------
// A (16x32 bf16) from fp32 row-major: lane[3:0]=M row, lane[4] selects K-halves.
// 4x b128 loads + hardware f32->bf16 cvt.
__device__ __forceinline__ v16bf load_a_f32(const float* __restrict__ A, int lda, int lane) {
  const float* p = A + (size_t)(lane & 15) * lda + ((lane >> 4) << 3);
  f4v f0 = *(const f4v*)(p);
  f4v f1 = *(const f4v*)(p + 4);
  f4v f2 = *(const f4v*)(p + 16);
  f4v f3 = *(const f4v*)(p + 20);
  v16bf a;
#pragma unroll
  for (int e = 0; e < 4; ++e) {
    a[e]      = (__bf16)f0[e];
    a[4 + e]  = (__bf16)f1[e];
    a[8 + e]  = (__bf16)f2[e];
    a[12 + e] = (__bf16)f3[e];
  }
  return a;
}
// B (32x16 bf16) from pre-transposed [N][K] bf16: lane[3:0]=N col, lane[4]=K group.
// 2x b128 loads, bit-reinterpreted.
__device__ __forceinline__ v16bf load_b_u16(const unsigned short* __restrict__ Bt, int ldb, int lane) {
  const unsigned short* p = Bt + (size_t)(lane & 15) * ldb + ((lane >> 4) << 3);
  union { u4v u; v8bf b; } lo, hi;
  lo.u = *(const u4v*)(p);
  hi.u = *(const u4v*)(p + 16);
  v16bf b;
#pragma unroll
  for (int e = 0; e < 8; ++e) { b[e] = lo.b[e]; b[8 + e] = hi.b[e]; }
  return b;
}

// ---- projection GEMM: C[M,N] = A[M,K] * Bt^T + bias -----------------------
// Each wave computes a 16x64 strip: one A fragment feeds 4 WMMAs per K-step.
// MODE 0: plain store at ldc. MODE 1: gated-half column interleave into DEXP.
template <int MODE, int HALFSEL>
__global__ void gemm_proj(const float* __restrict__ A, int lda,
                          const unsigned short* __restrict__ Bt, int ldb,
                          float* __restrict__ C, int ldc,
                          const float* __restrict__ bias,
                          int M, int N, int K) {
  int lane = threadIdx.x & 31;
  int wave = threadIdx.x >> 5;
  int nstrips_n = N >> 6;                 // strips of 4 n-tiles
  int strips = (M >> 4) * nstrips_n;
  int strip = blockIdx.x * 8 + wave;
  if (strip >= strips) return;
  int tm = strip / nstrips_n, sn = strip % nstrips_n;
  v8f acc[4] = {{}, {}, {}, {}};
  for (int kk = 0; kk < K; kk += 32) {
    v16bf a = load_a_f32(A + (size_t)(tm * 16) * lda + kk, lda, lane);
#pragma unroll
    for (int j = 0; j < 4; ++j) {
      v16bf b = load_b_u16(Bt + (size_t)(sn * 64 + j * 16) * ldb + kk, ldb, lane);
      acc[j] = __builtin_amdgcn_wmma_f32_16x16x32_bf16(false, a, false, b, (short)0, acc[j], false, false);
    }
  }
  int rbase = tm * 16 + ((lane >> 4) << 3);
#pragma unroll
  for (int j = 0; j < 4; ++j) {
    int col = sn * 64 + j * 16 + (lane & 15);
    float bv = bias[col];
#pragma unroll
    for (int r = 0; r < 8; ++r) {
      float v = acc[j][r] + bv;
      int row = rbase + r;
      if (MODE == 0) {
        C[(size_t)row * ldc + col] = v;
      } else {  // out col = head*64 + HALFSEL*32 + (col%32)
        int oc = ((col >> 5) << 6) + (HALFSEL << 5) + (col & 31);
        C[(size_t)row * DEXP + oc] = v;
      }
    }
  }
}

// ---- scores: per (b,h) S = (Qp Qp^T) * inv_t, M=N=1024, K=32 --------------
__global__ void gemm_scores(const float* __restrict__ Qp,
                            const unsigned short* __restrict__ QpB,
                            float* __restrict__ attn, float inv_t) {
  int lane = threadIdx.x & 31;
  int wave = threadIdx.x >> 5;
  int bh = blockIdx.y;
  int b = bh >> 3, h = bh & 7;
  int boff = b * DQK + h * DATT;
  int strip = blockIdx.x * 8 + wave;      // 1024 strips (64 tm x 16 sn)
  int tm = strip >> 4, sn = strip & 15;
  const int ld = BS * DQK;                // 2048
  v16bf a = load_a_f32(Qp + boff + (size_t)(tm * 16) * ld, ld, lane);
  v8f acc[4] = {{}, {}, {}, {}};
#pragma unroll
  for (int j = 0; j < 4; ++j) {
    v16bf bfr = load_b_u16(QpB + boff + (size_t)(sn * 64 + j * 16) * ld, ld, lane);
    acc[j] = __builtin_amdgcn_wmma_f32_16x16x32_bf16(false, a, false, bfr, (short)0, acc[j], false, false);
  }
  float* Cb = attn + (size_t)bh * LSEQ * LSEQ;
  int rbase = tm * 16 + ((lane >> 4) << 3);
#pragma unroll
  for (int j = 0; j < 4; ++j) {
    int col = sn * 64 + j * 16 + (lane & 15);
#pragma unroll
    for (int r = 0; r < 8; ++r)
      Cb[(size_t)(rbase + r) * LSEQ + col] = acc[j][r] * inv_t;
  }
}

// ---- row softmax over 1024, in place --------------------------------------
__global__ void softmax_rows(float* __restrict__ attn) {
  __shared__ float red[256];
  float* p = attn + (size_t)blockIdx.x * LSEQ;
  int t = threadIdx.x;
  float vals[4];
  float m = -3.4e38f;
#pragma unroll
  for (int i = 0; i < 4; ++i) { vals[i] = p[t + i * 256]; m = fmaxf(m, vals[i]); }
  red[t] = m; __syncthreads();
  for (int s = 128; s > 0; s >>= 1) { if (t < s) red[t] = fmaxf(red[t], red[t + s]); __syncthreads(); }
  m = red[0]; __syncthreads();
  float sum = 0.0f;
#pragma unroll
  for (int i = 0; i < 4; ++i) { vals[i] = __expf(vals[i] - m); sum += vals[i]; }
  red[t] = sum; __syncthreads();
  for (int s = 128; s > 0; s >>= 1) { if (t < s) red[t] += red[t + s]; __syncthreads(); }
  float inv = 1.0f / red[0];
#pragma unroll
  for (int i = 0; i < 4; ++i) p[t + i * 256] = vals[i] * inv;
}

// ---- attn @ Vc (per b,h), fused U-gate ------------------------------------
// One wave per 16x64 strip (full head dim), 4 WMMAs per K-step.
__global__ void gemm_attnv(const float* __restrict__ attn,
                           const unsigned short* __restrict__ VcT,
                           const float* __restrict__ Uc,
                           float* __restrict__ AG) {
  int lane = threadIdx.x & 31;
  int wave = threadIdx.x >> 5;
  int bh = blockIdx.y;
  int b = bh >> 3, h = bh & 7;
  int tm = blockIdx.x * 8 + wave;         // 64 strips (one per 16 rows)
  const float* Ab = attn + (size_t)bh * LSEQ * LSEQ + (size_t)(tm * 16) * LSEQ;
  const unsigned short* Bb = VcT + (size_t)bh * HD * LSEQ;
  v8f acc[4] = {{}, {}, {}, {}};
  for (int kk = 0; kk < LSEQ; kk += 32) {
    v16bf a = load_a_f32(Ab + kk, LSEQ, lane);
#pragma unroll
    for (int j = 0; j < 4; ++j) {
      v16bf bfr = load_b_u16(Bb + (size_t)(j * 16) * LSEQ + kk, LSEQ, lane);
      acc[j] = __builtin_amdgcn_wmma_f32_16x16x32_bf16(false, a, false, bfr, (short)0, acc[j], false, false);
    }
  }
  int rbase = tm * 16 + ((lane >> 4) << 3);
#pragma unroll
  for (int j = 0; j < 4; ++j) {
    int n = j * 16 + (lane & 15);
#pragma unroll
    for (int r = 0; r < 8; ++r) {
      int q = rbase + r;
      size_t idx = ((size_t)q * BS + b) * DEXP + h * HD + n;
      AG[idx] = acc[j][r] * Uc[idx];
    }
  }
}

// ---- helpers --------------------------------------------------------------
__global__ void wt_transpose(const float* __restrict__ W, unsigned short* __restrict__ Wt,
                             int K, int N) {  // W[K][N] -> Wt[N][K] bf16
  int i = blockIdx.x * 256 + threadIdx.x;
  if (i >= K * N) return;
  int n = i / K, k = i % K;
  Wt[i] = f2us(W[(size_t)k * N + n]);
}
__global__ void cvt_bf16(const float* __restrict__ X, unsigned short* __restrict__ Y, int n) {
  int i = blockIdx.x * 256 + threadIdx.x;
  if (i < n) Y[i] = f2us(X[i]);
}
__global__ void silu_inplace(float* __restrict__ X, int n) {
  int i = blockIdx.x * 256 + threadIdx.x;
  if (i < n) { float x = X[i]; X[i] = x / (1.0f + __expf(-x)); }
}
// VcT[bh][n=0..63][k=0..1023] = bf16(Vc[k, b, h*64+n])
__global__ void vct_build(const float* __restrict__ Vc, unsigned short* __restrict__ VcT) {
  int i = blockIdx.x * 256 + threadIdx.x;  // 4194304
  int k = i & (LSEQ - 1);
  int n = (i >> 10) & (HD - 1);
  int bh = i >> 16;
  int b = bh >> 3, h = bh & 7;
  VcT[i] = f2us(Vc[((size_t)k * BS + b) * DEXP + h * HD + n]);
}
__global__ void dwconv5(const float* __restrict__ AG, const float* __restrict__ Wc,
                        float* __restrict__ Y) {
  int i = blockIdx.x * 256 + threadIdx.x;  // 4194304
  int c = i & 511;
  int b = (i >> 9) & 7;
  int q = i >> 12;
  int ii = q >> 5, jj = q & 31;
  float s = 0.0f;
#pragma unroll
  for (int di = 0; di < 5; ++di) {
    int y0 = ii + di - 2;
    if ((unsigned)y0 >= 32u) continue;
#pragma unroll
    for (int dj = 0; dj < 5; ++dj) {
      int x0 = jj + dj - 2;
      if ((unsigned)x0 >= 32u) continue;
      s += AG[((size_t)(y0 * 32 + x0) * BS + b) * DEXP + c] * Wc[c * 25 + di * 5 + dj];
    }
  }
  Y[i] = s;
}

extern "C" void kernel_launch(void* const* d_in, const int* in_sizes, int n_in,
                              void* d_out, int out_size, void* d_ws, size_t ws_size,
                              hipStream_t stream) {
  (void)in_sizes; (void)n_in; (void)out_size; (void)ws_size;
  const float* Q   = (const float*)d_in[0];
  /* d_in[1] = K, discarded (use_linear path) */
  const float* V   = (const float*)d_in[2];
  const float* U   = (const float*)d_in[3];
  const float* Wqk = (const float*)d_in[4];
  const float* bqk = (const float*)d_in[5];
  const float* Wv1 = (const float*)d_in[6];
  const float* bv1 = (const float*)d_in[7];
  const float* Wv2 = (const float*)d_in[8];
  const float* bv2 = (const float*)d_in[9];
  const float* Wu1 = (const float*)d_in[10];
  const float* bu1 = (const float*)d_in[11];
  const float* Wu2 = (const float*)d_in[12];
  const float* bu2 = (const float*)d_in[13];
  const float* Wc  = (const float*)d_in[14];
  const float* Wp  = (const float*)d_in[15];
  const float* bp  = (const float*)d_in[16];

  float* out  = (float*)d_out;
  float* attn = out + (size_t)MROWS * DQK;  // [BS][NH][L][L]

  char* ws = (char*)d_ws;
  size_t off = 0;
#define WSALLOC(ty, name, bytes) ty* name = (ty*)(ws + off); off = (off + (size_t)(bytes) + 255) & ~(size_t)255
  WSALLOC(float, Qp,  (size_t)MROWS * DQK * 4);
  WSALLOC(float, Vc,  (size_t)MROWS * DEXP * 4);
  WSALLOC(float, Uc,  (size_t)MROWS * DEXP * 4);
  WSALLOC(float, AG,  (size_t)MROWS * DEXP * 4);
  WSALLOC(float, Yb,  (size_t)MROWS * DEXP * 4);
  WSALLOC(unsigned short, QpB,  (size_t)MROWS * DQK * 2);
  WSALLOC(unsigned short, VcT,  (size_t)BS * NH * HD * LSEQ * 2);
  WSALLOC(unsigned short, WqkT, 256 * 256 * 2);
  WSALLOC(unsigned short, Wv1T, 256 * 128 * 2);
  WSALLOC(unsigned short, Wv2T, 256 * 128 * 2);
  WSALLOC(unsigned short, Wu1T, 256 * 128 * 2);
  WSALLOC(unsigned short, Wu2T, 256 * 128 * 2);
  WSALLOC(unsigned short, WpT,  256 * 512 * 2);
#undef WSALLOC

  dim3 blk(256);
  // weight transposes -> bf16 [N][K]
  wt_transpose<<<(256 * 256) / 256, blk, 0, stream>>>(Wqk, WqkT, 256, 256);
  wt_transpose<<<(128 * 256) / 256, blk, 0, stream>>>(Wv1, Wv1T, 128, 256);
  wt_transpose<<<(128 * 256) / 256, blk, 0, stream>>>(Wv2, Wv2T, 128, 256);
  wt_transpose<<<(128 * 256) / 256, blk, 0, stream>>>(Wu1, Wu1T, 128, 256);
  wt_transpose<<<(128 * 256) / 256, blk, 0, stream>>>(Wu2, Wu2T, 128, 256);
  wt_transpose<<<(512 * 256) / 256, blk, 0, stream>>>(Wp, WpT, 512, 256);

  // Qp = Q @ Wqk + bqk  (8192x256x256); strips = 512*4 = 2048 -> 256 blocks
  gemm_proj<0, 0><<<256, blk, 0, stream>>>(Q, DQK, WqkT, 256, Qp, DQK, bqk, MROWS, DQK, DQK);
  cvt_bf16<<<(MROWS * DQK) / 256, blk, 0, stream>>>(Qp, QpB, MROWS * DQK);

  // gated half projections (interleaved columns), then SiLU
  gemm_proj<1, 0><<<256, blk, 0, stream>>>(V,           DVU, Wv1T, 128, Vc, 0, bv1, MROWS, 256, 128);
  gemm_proj<1, 1><<<256, blk, 0, stream>>>(V + HALF_IN, DVU, Wv2T, 128, Vc, 0, bv2, MROWS, 256, 128);
  gemm_proj<1, 0><<<256, blk, 0, stream>>>(U,           DVU, Wu1T, 128, Uc, 0, bu1, MROWS, 256, 128);
  gemm_proj<1, 1><<<256, blk, 0, stream>>>(U + HALF_IN, DVU, Wu2T, 128, Uc, 0, bu2, MROWS, 256, 128);
  silu_inplace<<<(MROWS * DEXP) / 256, blk, 0, stream>>>(Vc, MROWS * DEXP);
  silu_inplace<<<(MROWS * DEXP) / 256, blk, 0, stream>>>(Uc, MROWS * DEXP);

  // scores + softmax (attn materialized in d_out)
  gemm_scores<<<dim3(128, BS * NH), blk, 0, stream>>>(Qp, QpB, attn, 0.17677669529663687f);
  softmax_rows<<<BS * NH * LSEQ, blk, 0, stream>>>(attn);

  // attn @ Vc with fused U-gate
  vct_build<<<(BS * NH * HD * LSEQ) / 256, blk, 0, stream>>>(Vc, VcT);
  gemm_attnv<<<dim3(8, BS * NH), blk, 0, stream>>>(attn, VcT, Uc, AG);

  // depthwise 5x5 conv over 32x32 grid, then final projection
  dwconv5<<<(MROWS * DEXP) / 256, blk, 0, stream>>>(AG, Wc, Yb);
  gemm_proj<0, 0><<<256, blk, 0, stream>>>(Yb, DEXP, WpT, 512, out, DQK, bp, MROWS, DQK, DEXP);
}
